// SpatialLearning_87634512708238
// MI455X (gfx1250) — matrix operand order repaired and checked by reading, not compiled, not run on hardware
//
#include <hip/hip_runtime.h>

// Problem dims
#define N_  32
#define P_  16
#define C_  256
#define S_  128
#define H_  4
#define DH_ 64
#define CLS_ 3000

typedef __bf16 bf16x16 __attribute__((ext_vector_type(16)));
typedef float f32x8    __attribute__((ext_vector_type(8)));

union Frag { bf16x16 v; uint4 q[2]; unsigned short h[16]; };
union Acc  { f32x8 v; float f[8]; };

__device__ __forceinline__ unsigned short f2bf(float f) {
  unsigned int u = __float_as_uint(f);
  unsigned int r = u + 0x7FFFu + ((u >> 16) & 1u);   // round-to-nearest-even
  return (unsigned short)(r >> 16);
}
__device__ __forceinline__ float bfinf() { return -3.4e38f; }

#define WMMA_BF16(accu, fa, fb)                                           \
  (accu).v = __builtin_amdgcn_wmma_f32_16x16x32_bf16(                     \
      false, (fa).v, false, (fb).v, (short)0, (accu).v, false, false)

// ---- fragment loaders -------------------------------------------------
// A (16x32 bf16, M rows): lane m = lane&15; K = 16*(idx>>3) + 8*(lane>>4) + (idx&7)
__device__ __forceinline__ void loadA_g32(Frag &fr, const float* __restrict__ A,
                                          int ld, int m0, int k0, int lane) {
  const float* p = A + (m0 + (lane & 15)) * ld + k0 + ((lane >> 4) << 3);
  float4 v0 = *(const float4*)(p);
  float4 v1 = *(const float4*)(p + 4);
  float4 v2 = *(const float4*)(p + 16);
  float4 v3 = *(const float4*)(p + 20);
  fr.h[0]=f2bf(v0.x); fr.h[1]=f2bf(v0.y); fr.h[2]=f2bf(v0.z); fr.h[3]=f2bf(v0.w);
  fr.h[4]=f2bf(v1.x); fr.h[5]=f2bf(v1.y); fr.h[6]=f2bf(v1.z); fr.h[7]=f2bf(v1.w);
  fr.h[8]=f2bf(v2.x); fr.h[9]=f2bf(v2.y); fr.h[10]=f2bf(v2.z); fr.h[11]=f2bf(v2.w);
  fr.h[12]=f2bf(v3.x); fr.h[13]=f2bf(v3.y); fr.h[14]=f2bf(v3.z); fr.h[15]=f2bf(v3.w);
}
__device__ __forceinline__ void loadA_lds(Frag &fr, const unsigned short* S,
                                          int ld, int m0, int k0, int lane) {
  const unsigned short* p = S + (m0 + (lane & 15)) * ld + k0 + ((lane >> 4) << 3);
  fr.q[0] = *(const uint4*)(p);        // idx 0..7  (K .. K+7)
  fr.q[1] = *(const uint4*)(p + 16);   // idx 8..15 (K+16 .. K+23)
}
// B (32x16 bf16, K x N): lane n = lane&15; K = 16*(idx>>3) + 8*(lane>>4) + (idx&7)
__device__ __forceinline__ void loadB_lds(Frag &fr, const unsigned short* S,
                                          int ld, int n0, int k0, int lane) {
  const unsigned short* p = S + (n0 + (lane & 15)) * ld + k0 + ((lane >> 4) << 3);
  fr.q[0] = *(const uint4*)(p);
  fr.q[1] = *(const uint4*)(p + 16);
}
// B gathered from f32 row-major [K][ldB] (for fc), with N bound check
__device__ __forceinline__ void loadB_g32_scat(Frag &fr, const float* __restrict__ B,
                                               int ldB, int n0, int k0, int lane, int nmax) {
  int n = n0 + (lane & 15);
  int kk = k0 + ((lane >> 4) << 3);
  bool ok = (n < nmax);
#pragma unroll
  for (int j = 0; j < 16; ++j)
    fr.h[j] = f2bf(ok ? B[(kk + j + (j >= 8 ? 8 : 0)) * ldB + n] : 0.f);
}
// store D (f32 16x16: M = r + 8*(lane>>4), N = lane&15) transposed as bf16 S[N][M]
__device__ __forceinline__ void storeD_T(const Acc &a, unsigned short* S,
                                         int ld, int m0, int n0, int lane) {
  union { uint4 q; unsigned short h[8]; } t;
#pragma unroll
  for (int r = 0; r < 8; ++r) t.h[r] = f2bf(a.f[r]);
  *(uint4*)(S + (n0 + (lane & 15)) * ld + m0 + ((lane >> 4) << 3)) = t.q;
}
// store D natural as bf16 S[M][N]
__device__ __forceinline__ void storeD_N(const Acc &a, unsigned short* S,
                                         int ld, int m0, int n0, int lane) {
  int n = n0 + (lane & 15);
  int mb = m0 + ((lane >> 4) << 3);
#pragma unroll
  for (int r = 0; r < 8; ++r) S[(mb + r) * ld + n] = f2bf(a.f[r]);
}

// ---- LDS layout (bytes), phase-overlapped, total 320KB ----------------
// [0..196608)   phase A : Xst  [128][768] bf16
// [0..65536)    phase B : Tst  [128][256] bf16 (t_all transposed)
// [0..32768)    phase C : Attn [128][128] bf16
// [32768..33280) colsum f32[128] ; [33280..) idx
// [65536..131072)  Qt [128][256] bf16
// [131072..196608) Kt [128][256] bf16
// [196608..262144) Vn [256][128] bf16 (natural)
// [262144..327680) Ot [128][256] bf16 (o transposed)
#define SMEM_BYTES 327680

__global__ __launch_bounds__(256) void fused_attn_kernel(
    const float* __restrict__ tf, const float* __restrict__ ts,
    const float* __restrict__ tl, const float* __restrict__ decay_w,
    const float* __restrict__ qkv_w, const float* __restrict__ out_w,
    float* __restrict__ out_wpv, float* __restrict__ out_sel) {
  extern __shared__ char smem[];
  unsigned short* Xst = (unsigned short*)(smem);
  unsigned short* Tst = (unsigned short*)(smem);
  unsigned short* At  = (unsigned short*)(smem);
  float* colsum       = (float*)(smem + 32768);
  int*   idxsh        = (int*)(smem + 33280);
  unsigned short* Qt  = (unsigned short*)(smem + 65536);
  unsigned short* Kt  = (unsigned short*)(smem + 131072);
  unsigned short* Vn  = (unsigned short*)(smem + 196608);
  unsigned short* Ot  = (unsigned short*)(smem + 262144);

  const int blk = blockIdx.x;
  const int n = blk / P_, p = blk % P_;
  const int tid = threadIdx.x, wave = tid >> 5, lane = tid & 31;
  const int np = n * P_ + p;

  const float* tfb = tf + (size_t)np * C_ * S_;
  const float* tsb = ts + (size_t)np * C_ * S_;
  const float* tlb = tl + (size_t)np * C_ * S_;

  // ---------------- Phase A0: stage x_cat -> Xst[s][i] (bf16) ----------
  for (int e = tid; e < 768 * (S_ / 4); e += 256) {
    int i = e >> 5;           // 0..767
    int s = (e & 31) << 2;    // 0..124 step 4
    const float* sp = (i < 256 ? tfb : (i < 512 ? tsb : tlb)) + (i & 255) * S_ + s;
    float4 v = *(const float4*)sp;
    Xst[(s + 0) * 768 + i] = f2bf(v.x);
    Xst[(s + 1) * 768 + i] = f2bf(v.y);
    Xst[(s + 2) * 768 + i] = f2bf(v.z);
    Xst[(s + 3) * 768 + i] = f2bf(v.w);
  }
  __syncthreads();

  // ---------------- Phase A1: GEMM1  t_all = decay_w[p] * x_cat --------
  // wave owns rows [32*wave, 32*wave+32), all 128 cols; t_all kept in regs.
  Acc tA[2][8];
#pragma unroll
  for (int tm = 0; tm < 2; ++tm)
#pragma unroll
    for (int tn = 0; tn < 8; ++tn)
#pragma unroll
      for (int r = 0; r < 8; ++r) tA[tm][tn].f[r] = 0.f;

  const float* W1 = decay_w + (size_t)p * C_ * 768;
  for (int k0 = 0; k0 < 768; k0 += 32) {
    __builtin_prefetch(W1 + (32 * wave) * 768 + k0 + 64, 0, 0);
    Frag a0, a1;
    loadA_g32(a0, W1, 768, 32 * wave, k0, lane);
    loadA_g32(a1, W1, 768, 32 * wave + 16, k0, lane);
#pragma unroll
    for (int tn = 0; tn < 8; ++tn) {
      Frag b; loadB_lds(b, Xst, 768, 16 * tn, k0, lane);
      WMMA_BF16(tA[0][tn], a0, b);
      WMMA_BF16(tA[1][tn], a1, b);
    }
  }
  __syncthreads();   // all waves done reading Xst
#pragma unroll
  for (int tm = 0; tm < 2; ++tm)
#pragma unroll
    for (int tn = 0; tn < 8; ++tn)
      storeD_T(tA[tm][tn], Tst, 256, 32 * wave + 16 * tm, 16 * tn, lane);
  __syncthreads();

  // ---------------- Phase B: GEMM2  qkv = qkv_w[p] * t_all -------------
  const float* W2 = qkv_w + (size_t)p * 768 * C_;
  for (int pass = 0; pass < 3; ++pass) {
    Acc acc[2][8];
#pragma unroll
    for (int tm = 0; tm < 2; ++tm)
#pragma unroll
      for (int tn = 0; tn < 8; ++tn)
#pragma unroll
        for (int r = 0; r < 8; ++r) acc[tm][tn].f[r] = 0.f;
    int mbase = pass * 256 + 32 * wave;
    for (int k0 = 0; k0 < 256; k0 += 32) {
      __builtin_prefetch(W2 + mbase * 256 + k0 + 64, 0, 0);
      Frag a0, a1;
      loadA_g32(a0, W2, 256, mbase, k0, lane);
      loadA_g32(a1, W2, 256, mbase + 16, k0, lane);
#pragma unroll
      for (int tn = 0; tn < 8; ++tn) {
        Frag b; loadB_lds(b, Tst, 256, 16 * tn, k0, lane);
        WMMA_BF16(acc[0][tn], a0, b);
        WMMA_BF16(acc[1][tn], a1, b);
      }
    }
#pragma unroll
    for (int tm = 0; tm < 2; ++tm)
#pragma unroll
      for (int tn = 0; tn < 8; ++tn) {
        int m0 = 32 * wave + 16 * tm;
        if (pass == 0)      storeD_T(acc[tm][tn], Qt, 256, m0, 16 * tn, lane);
        else if (pass == 1) storeD_T(acc[tm][tn], Kt, 256, m0, 16 * tn, lane);
        else                storeD_N(acc[tm][tn], Vn, 128, m0, 16 * tn, lane);
      }
  }
  __syncthreads();

  // ---------------- Phase C: per-head attention ------------------------
  float selAcc = 0.f;
  for (int h = 0; h < H_; ++h) {
    if (tid < 128) colsum[tid] = 0.f;
    __syncthreads();

    // dots[s,t] = sum_c q[c,s]*k[c,t] ; wave owns rows s = 16*wave..+15
    Acc d[8];
#pragma unroll
    for (int tn = 0; tn < 8; ++tn)
#pragma unroll
      for (int r = 0; r < 8; ++r) d[tn].f[r] = 0.f;
    const int s0 = 16 * wave;
    for (int k0 = 0; k0 < DH_; k0 += 32) {
      Frag a; loadA_lds(a, Qt, 256, s0, h * DH_ + k0, lane);
#pragma unroll
      for (int tn = 0; tn < 8; ++tn) {
        Frag b; loadB_lds(b, Kt, 256, 16 * tn, h * DH_ + k0, lane);
        WMMA_BF16(d[tn], a, b);
      }
    }
    // softmax over t (scale = dh^-0.5 = 0.125), rows live in lane-halves
#pragma unroll
    for (int r = 0; r < 8; ++r) {
      float m = bfinf();
#pragma unroll
      for (int tn = 0; tn < 8; ++tn) m = fmaxf(m, d[tn].f[r]);
#pragma unroll
      for (int mask = 1; mask < 16; mask <<= 1) m = fmaxf(m, __shfl_xor(m, mask, 32));
      float ssum = 0.f;
#pragma unroll
      for (int tn = 0; tn < 8; ++tn) {
        float e = __expf((d[tn].f[r] - m) * 0.125f);
        d[tn].f[r] = e; ssum += e;
      }
#pragma unroll
      for (int mask = 1; mask < 16; mask <<= 1) ssum += __shfl_xor(ssum, mask, 32);
      float inv = 1.f / ssum;
#pragma unroll
      for (int tn = 0; tn < 8; ++tn) d[tn].f[r] *= inv;
    }
    // column sums (ds_add_f32) + store attn bf16 At[s][t]
#pragma unroll
    for (int tn = 0; tn < 8; ++tn) {
      float cs = 0.f;
#pragma unroll
      for (int r = 0; r < 8; ++r) cs += d[tn].f[r];
      atomicAdd(&colsum[16 * tn + (lane & 15)], cs);
      int tcol = 16 * tn + (lane & 15);
      int srow = s0 + ((lane >> 4) << 3);
#pragma unroll
      for (int r = 0; r < 8; ++r) At[(srow + r) * 128 + tcol] = f2bf(d[tn].f[r]);
    }
    __syncthreads();

    // argmax over t of colsum (first-index tie break)
    if (wave == 0) {
      float bv = bfinf(); int bi = 0;
      for (int t = lane; t < 128; t += 32) {
        float v = colsum[t];
        if (v > bv) { bv = v; bi = t; }
      }
      for (int mask = 16; mask; mask >>= 1) {
        float ov = __shfl_xor(bv, mask, 32);
        int   oi = __shfl_xor(bi, mask, 32);
        if (ov > bv || (ov == bv && oi < bi)) { bv = ov; bi = oi; }
      }
      if (lane == 0) *idxsh = bi;
    }
    __syncthreads();
    selAcc += tfb[tid * S_ + *idxsh];   // tid == channel c (0..255)

    // o[c,s] = sum_t v[c,t] * attn[s,t]; 8 waves: 4 row tiles x 2 col halves
    Acc ov[4];
#pragma unroll
    for (int tn = 0; tn < 4; ++tn)
#pragma unroll
      for (int r = 0; r < 8; ++r) ov[tn].f[r] = 0.f;
    const int mrel = 16 * (wave & 3);
    const int nb = 64 * (wave >> 2);
    for (int k0 = 0; k0 < 128; k0 += 32) {
      Frag a; loadA_lds(a, Vn, 128, h * DH_ + mrel, k0, lane);
#pragma unroll
      for (int tn = 0; tn < 4; ++tn) {
        Frag b; loadB_lds(b, At, 128, nb + 16 * tn, k0, lane);
        WMMA_BF16(ov[tn], a, b);
      }
    }
#pragma unroll
    for (int tn = 0; tn < 4; ++tn)
      storeD_T(ov[tn], Ot, 256, h * DH_ + mrel, nb + 16 * tn, lane);
    __syncthreads();
  }

  // ---------------- Phase D: out proj + residual + frame max -----------
  const float* W3 = out_w + (size_t)p * C_ * C_;
  Acc o2[2][8];
#pragma unroll
  for (int tm = 0; tm < 2; ++tm)
#pragma unroll
    for (int tn = 0; tn < 8; ++tn)
#pragma unroll
      for (int r = 0; r < 8; ++r) o2[tm][tn].f[r] = 0.f;
  for (int k0 = 0; k0 < 256; k0 += 32) {
    __builtin_prefetch(W3 + (32 * wave) * 256 + k0 + 64, 0, 0);
    Frag a0, a1;
    loadA_g32(a0, W3, 256, 32 * wave, k0, lane);
    loadA_g32(a1, W3, 256, 32 * wave + 16, k0, lane);
#pragma unroll
    for (int tn = 0; tn < 8; ++tn) {
      Frag b; loadB_lds(b, Ot, 256, 16 * tn, k0, lane);
      WMMA_BF16(o2[0][tn], a0, b);
      WMMA_BF16(o2[1][tn], a1, b);
    }
  }
#pragma unroll
  for (int tm = 0; tm < 2; ++tm) {
    float vm[8];
#pragma unroll
    for (int r = 0; r < 8; ++r) vm[r] = bfinf();
#pragma unroll
    for (int tn = 0; tn < 8; ++tn)
#pragma unroll
      for (int r = 0; r < 8; ++r)
        vm[r] = fmaxf(vm[r], o2[tm][tn].f[r] + tA[tm][tn].f[r]);
#pragma unroll
    for (int r = 0; r < 8; ++r)
#pragma unroll
      for (int mask = 1; mask < 16; mask <<= 1)
        vm[r] = fmaxf(vm[r], __shfl_xor(vm[r], mask, 32));
    if ((lane & 15) == 0) {
      int half = lane >> 4;
#pragma unroll
      for (int r = 0; r < 8; ++r) {
        int c = 32 * wave + 16 * tm + 8 * half + r;
        out_wpv[p * (N_ * C_) + n * C_ + c] = vm[r];
      }
    }
  }
  out_sel[p * (N_ * C_) + n * C_ + tid] = selAcc;
}

// ---------------- Kernel 2: per-part BN (batch stats) + classifier -----
__global__ __launch_bounds__(256) void bn_fc_kernel(
    const float* __restrict__ wpv,   // [p][n][c]
    const float* __restrict__ gamma, const float* __restrict__ beta,
    const float* __restrict__ fc,    // [C][CLS]
    float* __restrict__ cls) {       // [n][p][CLS]
  __shared__ unsigned short Pf[N_ * C_];   // pf bf16 [32][256]
  const int p = blockIdx.x, chunk = blockIdx.y;
  const int tid = threadIdx.x, wave = tid >> 5, lane = tid & 31;

  {  // batch stats over n (biased var, matching torch BN training mode)
    const int c = tid;
    float vals[N_], s = 0.f, sq = 0.f;
#pragma unroll
    for (int nn = 0; nn < N_; ++nn) {
      float v = wpv[p * (N_ * C_) + nn * C_ + c];
      vals[nn] = v; s += v; sq += v * v;
    }
    float mean = s * (1.f / N_);
    float var = sq * (1.f / N_) - mean * mean;
    float inv = rsqrtf(var + 1e-5f) * gamma[p * C_ + c];
    float b = beta[p * C_ + c];
#pragma unroll
    for (int nn = 0; nn < N_; ++nn)
      Pf[nn * C_ + c] = f2bf((vals[nn] - mean) * inv + b);
  }
  __syncthreads();

  const int NT = (CLS_ + 15) / 16;          // 188 N tiles
  int tn0 = chunk * 24, tn1 = min(NT, tn0 + 24);
  int ntiles = (tn1 - tn0) * 2;
  for (int t = wave; t < ntiles; t += 8) {
    int tm = t & 1, tn = tn0 + (t >> 1);
    Acc acc;
#pragma unroll
    for (int r = 0; r < 8; ++r) acc.f[r] = 0.f;
    for (int k0 = 0; k0 < C_; k0 += 32) {
      Frag a; loadA_lds(a, Pf, C_, 16 * tm, k0, lane);
      Frag b; loadB_g32_scat(b, fc, CLS_, 16 * tn, k0, lane, CLS_);
      WMMA_BF16(acc, a, b);
    }
    int kout = 16 * tn + (lane & 15);
    if (kout < CLS_) {
      int nb = 16 * tm + ((lane >> 4) << 3);
#pragma unroll
      for (int r = 0; r < 8; ++r)
        cls[(size_t)(nb + r) * (P_ * CLS_) + p * CLS_ + kout] = acc.f[r];
    }
  }
}

extern "C" void kernel_launch(void* const* d_in, const int* in_sizes, int n_in,
                              void* d_out, int out_size, void* d_ws, size_t ws_size,
                              hipStream_t stream) {
  const float* tf      = (const float*)d_in[0];
  const float* ts      = (const float*)d_in[1];
  const float* tl      = (const float*)d_in[2];
  const float* decay_w = (const float*)d_in[3];
  const float* qkv_w   = (const float*)d_in[4];
  const float* out_w   = (const float*)d_in[5];
  const float* gamma   = (const float*)d_in[6];
  const float* beta    = (const float*)d_in[7];
  const float* fc      = (const float*)d_in[8];

  float* out_cls = (float*)d_out;                  // [n][p][CLS]
  float* out_wpv = out_cls + (size_t)N_ * P_ * CLS_;  // [p][n][c]
  float* out_sel = out_wpv + (size_t)P_ * N_ * C_;    // [p][n][c]

  fused_attn_kernel<<<N_ * P_, 256, SMEM_BYTES, stream>>>(
      tf, ts, tl, decay_w, qkv_w, out_w, out_wpv, out_sel);
  bn_fc_kernel<<<dim3(P_, 8), 256, 0, stream>>>(out_wpv, gamma, beta, fc, out_cls);
}
